// BiGNN_53695681134823
// MI455X (gfx1250) — matrix operand order repaired and checked by reading, not compiled, not run on hardware
//
#include <hip/hip_runtime.h>
#include <hip/hip_bf16.h>
#include <stdint.h>

#define B_DIM   2
#define N_LR    8192
#define N_HR    16384
#define CCH     32
#define NS      64
#define CHUNK   2048
#define NCHUNK  (N_HR / CHUNK)
#define WAVES   4
#define THREADS (WAVES * 32)

typedef __bf16 bf16;
typedef bf16  v16bf __attribute__((ext_vector_type(16)));
typedef float v8f   __attribute__((ext_vector_type(8)));
typedef unsigned int v4u __attribute__((ext_vector_type(4)));
typedef int  v8i __attribute__((ext_vector_type(8)));
typedef int  v4i __attribute__((ext_vector_type(4)));

#if __has_builtin(__builtin_amdgcn_tensor_load_to_lds) && __has_builtin(__builtin_amdgcn_s_wait_tensorcnt)
#define HAVE_TDM 1
#else
#define HAVE_TDM 0
#endif

union Frag { v16bf v; uint4 q[2]; unsigned short h[16]; };

__device__ inline unsigned short f2bf(float f) {
  union { float f; unsigned u; } x; x.f = f;
  unsigned r = x.u + 0x7fffu + ((x.u >> 16) & 1u);   // round-to-nearest-even
  return (unsigned short)(r >> 16);
}
__device__ inline unsigned packbf(float a, float b) {
  return (unsigned)f2bf(a) | ((unsigned)f2bf(b) << 16);
}
__device__ inline unsigned ballot32(bool p) {
#if __has_builtin(__builtin_amdgcn_ballot_w32)
  return __builtin_amdgcn_ballot_w32(p);
#else
  return (unsigned)__ballot(p ? 1 : 0);
#endif
}

__global__ __launch_bounds__(THREADS) void bignn_fused(
    const int* __restrict__ lr_idx, const int* __restrict__ hr_idx,
    const float* __restrict__ lr_feat, const float* __restrict__ hr_feat,
    const float* __restrict__ W1, const float* __restrict__ b1,
    const float* __restrict__ W2, const float* __restrict__ b2,
    float* __restrict__ out)
{
  // chunk buffer (24 KB) and grouped-feature tiles (32 KB) are time-disjoint
  __shared__ __align__(16) unsigned char s_union[32768];
  int*            s_chunk = (int*)s_union;                 // CHUNK * 3 ints
  float*          s_cxyz  = (float*)s_union;               // converted in place
  unsigned short* s_G     = (unsigned short*)s_union;      // WAVES * 64*64 bf16
  __shared__ int s_nidx[WAVES][NS];
  // weights pre-swizzled into B-fragment order: [frag][lane][elem] -> 32B/lane
  __shared__ __align__(16) unsigned short s_W1f[4 * 32 * 16];  // (ks,nt) x lane x 16
  __shared__ __align__(16) unsigned short s_W2f[2 * 32 * 16];  // (nt)    x lane x 16
  __shared__ float s_b1[32], s_b2[32];
  __shared__ __align__(16) unsigned short s_H1[WAVES][16 * 32];

  const int tid  = threadIdx.x;
  const int lane = tid & 31;
  const int w    = tid >> 5;
  const int blocks_per_b = N_LR / WAVES;
  const int b = blockIdx.x / blocks_per_b;
  const int q = (blockIdx.x % blocks_per_b) * WAVES + w;

  // ---- phase 0: weights/biases -> LDS, already in fragment layout ----
  // frag element i of lane l corresponds to K = kbase + 16*(l>>4) + i, N = nt*16 + (l&15)
  for (int e = tid; e < 4 * 32 * 16; e += THREADS) {
    int i  = e & 15;
    int l  = (e >> 4) & 31;
    int fi = e >> 9;                       // ks*2 + nt
    int ks = fi >> 1, nt = fi & 1;
    int k = ks * 32 + (l >> 4) * 16 + i;
    int n = nt * 16 + (l & 15);
    s_W1f[e] = (k < 35) ? f2bf(W1[k * 32 + n]) : (unsigned short)0;
  }
  for (int e = tid; e < 2 * 32 * 16; e += THREADS) {
    int i  = e & 15;
    int l  = (e >> 4) & 31;
    int nt = e >> 9;
    int k = (l >> 4) * 16 + i;
    int n = nt * 16 + (l & 15);
    s_W2f[e] = f2bf(W2[k * 32 + n]);
  }
  if (tid < 32) { s_b1[tid] = b1[tid]; s_b2[tid] = b2[tid]; }

  // query xyz: idx is (z,y,x); xyz = idx_rev*vox + off + 0.5*vox
  const int* qi = lr_idx + ((size_t)b * N_LR + q) * 3;
  const float qx = qi[2] * 0.4f + 0.2f;
  const float qy = qi[1] * 0.4f + (-40.0f + 0.2f);
  const float qz = qi[0] * 1.0f + (-3.0f + 0.5f);

  for (int s = lane; s < NS; s += 32) s_nidx[w][s] = 0;
  int cnt = 0;

  const int* hr_b = hr_idx + (size_t)b * N_HR * 3;

  // ---- phase 1: ball query, TDM-staged LDS chunks, first-NS ascending ----
  for (int c = 0; c < NCHUNK; ++c) {
    __syncthreads();                       // previous chunk fully consumed
#if HAVE_TDM
    if (w == 0) {                          // one wave drives the Tensor Data Mover
      unsigned long long ga = (unsigned long long)(uintptr_t)(hr_b + c * CHUNK * 3);
      unsigned lds = (unsigned)(uintptr_t)(void*)s_chunk;
      const unsigned NE = CHUNK * 3;       // 4-byte elements
      v4u g0;
      g0[0] = 1u;                                   // count=1, user descriptor
      g0[1] = lds;                                  // lds_addr
      g0[2] = (unsigned)ga;                         // global_addr[31:0]
      g0[3] = (unsigned)((ga >> 32) & 0x1ffffffu) | 0x80000000u; // addr[56:32] | type=2
      v8i g1;
      g1[0] = (int)(2u << 16);                      // data_size=4B
      g1[1] = (int)((NE & 0xffffu) << 16);          // tensor_dim0[15:0]
      g1[2] = (int)((NE >> 16) | (1u << 16));       // tensor_dim0[31:16] | tensor_dim1=1
      g1[3] = (int)(NE << 16);                      // tile_dim0
      g1[4] = 1;                                    // tile_dim1=1
      g1[5] = (int)NE;                              // tensor_dim0_stride lo
      g1[6] = 0; g1[7] = 0;
      v4i gz4 = {0, 0, 0, 0};
      v8i gz8 = {0, 0, 0, 0, 0, 0, 0, 0};
      __builtin_amdgcn_tensor_load_to_lds(g0, g1, gz4, gz4, gz8, 0);
      __builtin_amdgcn_s_wait_tensorcnt(0);
    }
#else
    for (int i = tid; i < CHUNK * 3; i += THREADS)
      s_chunk[i] = hr_b[(size_t)c * CHUNK * 3 + i];
#endif
    __syncthreads();

    // convert staged voxel indices to xyz in place, once per chunk
    // (each chunk is scanned by all 4 waves; don't redo cvt+fma per wave)
    for (int i = tid; i < CHUNK * 3; i += THREADS) {
      int comp = i - (i / 3) * 3;          // 0=z, 1=y, 2=x
      float sc = (comp == 0) ? 0.1f  : 0.05f;
      float of = (comp == 0) ? -2.95f : ((comp == 1) ? -39.975f : 0.025f);
      s_cxyz[i] = (float)s_chunk[i] * sc + of;
    }
    __syncthreads();

    if (cnt < NS) {
      for (int it = 0; it < CHUNK / 32; ++it) {
        const float* cp = s_cxyz + (it * 32 + lane) * 3;
        float dz = cp[0] - qz, dy = cp[1] - qy, dx = cp[2] - qx;
        bool pred = (dx * dx + dy * dy + dz * dz) <= 1.0f;
        unsigned m = ballot32(pred);
        if (pred) {
          int pos = cnt + __popc(m & ((1u << lane) - 1u));
          if (pos < NS) s_nidx[w][pos] = c * CHUNK + it * 32 + lane;
        }
        cnt += __popc(m);
        if (cnt >= NS) break;              // wave-uniform
      }
    }
  }
  __syncthreads();                         // release s_chunk region for s_G

  // pad empty slots with first valid neighbor (0 if none) — matches reference
  int nvalid = cnt < NS ? cnt : NS;
  int first  = s_nidx[w][0];
  for (int s = nvalid + lane; s < NS; s += 32) s_nidx[w][s] = first;

  // ---- phase 2: gather [rel | feat] as bf16 rows (64 x 64, K-padded) ----
  unsigned short* Gw = s_G + w * (64 * 64);
  for (int s = lane; s < NS; s += 32) {
    int h = s_nidx[w][s];
    const int*   hi3 = hr_idx  + ((size_t)b * N_HR + h) * 3;
    const float* fr  = hr_feat + ((size_t)b * N_HR + h) * CCH;
    __builtin_prefetch(fr, 0, 0);
    int iz = hi3[0], iy = hi3[1], ix = hi3[2];
    float t[36];
    t[0] = ix * 0.05f + 0.025f            - qx;
    t[1] = iy * 0.05f + (-40.0f + 0.025f) - qy;
    t[2] = iz * 0.1f  + (-3.0f + 0.05f)   - qz;
    t[35] = 0.0f;
    const float4* f4 = (const float4*)fr;
#pragma unroll
    for (int j = 0; j < 8; ++j) {
      float4 v = f4[j];
      t[3 + 4 * j + 0] = v.x; t[3 + 4 * j + 1] = v.y;
      t[3 + 4 * j + 2] = v.z; t[3 + 4 * j + 3] = v.w;
    }
    unsigned* row = (unsigned*)(Gw + s * 64);
#pragma unroll
    for (int p = 0; p < 18; ++p) row[p] = packbf(t[2 * p], t[2 * p + 1]);
#pragma unroll
    for (int p = 18; p < 32; ++p) row[p] = 0u;
  }
  __syncthreads();

  // ---- phase 3: MLP via WMMA + max-pool ----
  const int col = lane & 15;               // N (B/C) or M (A) position
  const int hi  = lane >> 4;               // lane half

  // B fragments: two ds_load_b128 each (pre-swizzled in phase 0)
  Frag B1f[2][2], B2f[2];
#pragma unroll
  for (int ks = 0; ks < 2; ++ks)
#pragma unroll
    for (int nt = 0; nt < 2; ++nt) {
      const unsigned short* p = s_W1f + ((ks * 2 + nt) * 32 + lane) * 16;
      B1f[ks][nt].q[0] = *(const uint4*)(p);
      B1f[ks][nt].q[1] = *(const uint4*)(p + 8);
    }
#pragma unroll
  for (int nt = 0; nt < 2; ++nt) {
    const unsigned short* p = s_W2f + (nt * 32 + lane) * 16;
    B2f[nt].q[0] = *(const uint4*)(p);
    B2f[nt].q[1] = *(const uint4*)(p + 8);
  }

  float vmax[2][8];
#pragma unroll
  for (int nt = 0; nt < 2; ++nt)
#pragma unroll
    for (int r = 0; r < 8; ++r) vmax[nt][r] = 0.0f;   // relu outputs are >= 0

  unsigned short* Hw = s_H1[w];

  for (int m = 0; m < 4; ++m) {
    // A frags: lane half selects K offset 0/16 within each 32-wide K step
    const unsigned short* Grow = Gw + (m * 16 + col) * 64;
    Frag a0, a1;
    a0.q[0] = *(const uint4*)(Grow +  0 + hi * 16);
    a0.q[1] = *(const uint4*)(Grow +  8 + hi * 16);
    a1.q[0] = *(const uint4*)(Grow + 32 + hi * 16);
    a1.q[1] = *(const uint4*)(Grow + 40 + hi * 16);

    v8f acc[2];
#pragma unroll
    for (int nt = 0; nt < 2; ++nt) {
      float bias = s_b1[nt * 16 + col];
#pragma unroll
      for (int r = 0; r < 8; ++r) acc[nt][r] = bias;
      acc[nt] = __builtin_amdgcn_wmma_f32_16x16x32_bf16(
          false, a0.v, false, B1f[0][nt].v, (short)0, acc[nt], false, false);
      acc[nt] = __builtin_amdgcn_wmma_f32_16x16x32_bf16(
          false, a1.v, false, B1f[1][nt].v, (short)0, acc[nt], false, false);
#pragma unroll
      for (int r = 0; r < 8; ++r) {
        float v = fmaxf(acc[nt][r], 0.0f);                 // relu -> v_max_num_f32
        Hw[(r + 8 * hi) * 32 + nt * 16 + col] = f2bf(v);   // C-layout -> row-major
      }
    }
    __syncthreads();

    Frag a2;                                // reload H1 tile in A layout
    const unsigned short* Hrow = Hw + col * 32;
    a2.q[0] = *(const uint4*)(Hrow +  0 + hi * 16);
    a2.q[1] = *(const uint4*)(Hrow +  8 + hi * 16);
#pragma unroll
    for (int nt = 0; nt < 2; ++nt) {
      v8f acc2;
      float bias = s_b2[nt * 16 + col];
#pragma unroll
      for (int r = 0; r < 8; ++r) acc2[r] = bias;
      acc2 = __builtin_amdgcn_wmma_f32_16x16x32_bf16(
          false, a2.v, false, B2f[nt].v, (short)0, acc2, false, false);
#pragma unroll
      for (int r = 0; r < 8; ++r)
        vmax[nt][r] = fmaxf(vmax[nt][r], fmaxf(acc2[r], 0.0f));
    }
    __syncthreads();
  }

  // ---- phase 4: finish max-pool, write [lr_feat | pooled] ----
  float* orow = out + ((size_t)b * N_LR + q) * 64;
  orow[lane] = lr_feat[((size_t)b * N_LR + q) * CCH + lane];
#pragma unroll
  for (int nt = 0; nt < 2; ++nt) {
    float r = vmax[nt][0];
#pragma unroll
    for (int j = 1; j < 8; ++j) r = fmaxf(r, vmax[nt][j]);
    float o = __shfl_xor(r, 16, 32);        // fold M halves (lane l <-> l+16)
    r = fmaxf(r, o);
    if (lane < 16) orow[32 + nt * 16 + lane] = r;
  }
}

extern "C" void kernel_launch(void* const* d_in, const int* in_sizes, int n_in,
                              void* d_out, int out_size, void* d_ws, size_t ws_size,
                              hipStream_t stream) {
  (void)in_sizes; (void)n_in; (void)out_size; (void)d_ws; (void)ws_size;
  const int*   lr_idx  = (const int*)d_in[0];
  const int*   hr_idx  = (const int*)d_in[1];
  const float* lr_feat = (const float*)d_in[2];
  const float* hr_feat = (const float*)d_in[3];
  const float* W1 = (const float*)d_in[4];
  const float* b1 = (const float*)d_in[5];
  const float* W2 = (const float*)d_in[6];
  const float* b2 = (const float*)d_in[7];
  float* out = (float*)d_out;

  dim3 grid(B_DIM * (N_LR / WAVES));
  bignn_fused<<<grid, THREADS, 0, stream>>>(lr_idx, hr_idx, lr_feat, hr_feat,
                                            W1, b1, W2, b2, out);
}
